// GRUController_14937896255907
// MI455X (gfx1250) — compile-verified, matrix operand order checked by reading, not run on hardware
//
#include <hip/hip_runtime.h>
#include <hip/hip_bf16.h>

#define NN 2048
#define TT 256
#define DZC 256
#define PTC 64
#define PP 512
#define HH 512
#define GG 1536   // 3*H

typedef __attribute__((ext_vector_type(16))) __bf16 v16bf;
typedef __attribute__((ext_vector_type(8)))  float  v8f;

// ---------------- wave32 reduction ----------------
__device__ __forceinline__ float wred32(float v) {
#pragma unroll
    for (int m = 16; m; m >>= 1) v += __shfl_xor(v, m, 32);
    return v;
}

// ---------------- CDNA5 async global->LDS copy (16B), ASYNCcnt-tracked ------
__device__ __forceinline__ void async_cp16(uint32_t ldsoff, const __bf16* g) {
    asm volatile("global_load_async_to_lds_b128 %0, %1, off"
                 :: "v"(ldsoff), "v"((unsigned long long)(uintptr_t)g)
                 : "memory");
}
#define LDSOFF(p) ((uint32_t)(uintptr_t)(p))

// ---------------- WMMA fragment load from row-major LDS tile ----------------
// 16-bit operand layout (ISA 7.12.2): lanes 0-15 hold K={0..7,16..23},
// lanes 16-31 hold K={8..15,24..31}. Two ds_read_b128 per fragment.
__device__ __forceinline__ v16bf frag_ld(const __bf16* rowp, int lane) {
    int kb = (lane & 16) ? 8 : 0;
    const uint4* p = (const uint4*)(rowp + kb);
    union { uint4 q[2]; v16bf v; } u;
    u.q[0] = p[0];   // K = kb .. kb+7
    u.q[1] = p[2];   // K = kb+16 .. kb+23
    return u.v;
}

// ================= Kernel 0: one-time fp32 -> bf16 weight conversion ========
__global__ void k_cvtw(const float* __restrict__ Wih, const float* __restrict__ Whh,
                       __bf16* __restrict__ wihb, __bf16* __restrict__ whhb) {
    const size_t E = (size_t)GG * PP;
    size_t base = ((size_t)blockIdx.x * 256 + threadIdx.x) * 4;
    if (base < E) {
#pragma unroll
        for (int j = 0; j < 4; ++j) wihb[base + j] = (__bf16)Wih[base + j];
    } else {
        base -= E;
#pragma unroll
        for (int j = 0; j < 4; ++j) whhb[base + j] = (__bf16)Whh[base + j];
    }
}

// ================= Kernel 1: split projection  az = z*Wz^T + b, at = tpe*Wt^T
__global__ void k_proj(const float* __restrict__ z, const float* __restrict__ tpe,
                       const float* __restrict__ W, const float* __restrict__ bp,
                       float* __restrict__ az, float* __restrict__ at) {
    int idx = blockIdx.x * 256 + threadIdx.x;
    const int totalA = NN * PP;
    if (idx < totalA) {
        int n = idx / PP, p = idx % PP;
        const float* wr = W + (size_t)p * (DZC + PTC);
        const float* zr = z + (size_t)n * DZC;
        float s = bp[p];
#pragma unroll 8
        for (int k = 0; k < DZC; ++k) s = fmaf(zr[k], wr[k], s);
        az[idx] = s;
    } else {
        int j = idx - totalA;
        if (j < TT * PP) {
            int t = j / PP, p = j % PP;
            const float* wr = W + (size_t)p * (DZC + PTC) + DZC;
            const float* tr = tpe + (size_t)t * PTC;
            float s = 0.0f;
#pragma unroll 8
            for (int k = 0; k < PTC; ++k) s = fmaf(tr[k], wr[k], s);
            at[j] = s;
        }
    }
}

// ================= Kernel 2: x = LN(GELU(az[n]+at[t])) -> bf16, one wave/row
__global__ void k_xln(const float* __restrict__ az, const float* __restrict__ at,
                      const float* __restrict__ lg, const float* __restrict__ lb,
                      __bf16* __restrict__ xg) {
    int wave = threadIdx.x >> 5, lane = threadIdx.x & 31;
    int row  = blockIdx.x * 8 + wave;          // row in [0, T*N), row = t*N + n
    int t = row / NN, n = row % NN;
    const float* a0 = az + (size_t)n * PP;
    const float* a1 = at + (size_t)t * PP;
    float v[16], s = 0.0f, s2 = 0.0f;
#pragma unroll
    for (int j = 0; j < 16; ++j) {
        int c = lane + 32 * j;
        float x = a0[c] + a1[c];
        x = 0.5f * x * (1.0f + erff(x * 0.70710678118654752f));   // exact GELU
        v[j] = x; s += x; s2 = fmaf(x, x, s2);
    }
    s = wred32(s); s2 = wred32(s2);
    float mean = s * (1.0f / PP);
    float var  = s2 * (1.0f / PP) - mean * mean;
    float rstd = rsqrtf(var + 1e-5f);
    __bf16* o = xg + (size_t)row * PP;
#pragma unroll
    for (int j = 0; j < 16; ++j) {
        int c = lane + 32 * j;
        o[c] = (__bf16)(((v[j] - mean) * rstd) * lg[c] + lb[c]);
    }
}

// ================= Kernel 3: gx = x @ W_ih^T + b_ih (bf16 out)
// grid = (GG/128, (T*N)/128); block = 256 (8 waves). M=N=128, K=512.
// Double-buffered LDS fed by async global->LDS copies (4 per wave per tile).
__global__ void k_gx(const __bf16* __restrict__ xg, const __bf16* __restrict__ wihb,
                     const float* __restrict__ bih, __bf16* __restrict__ gx) {
    __shared__ __align__(16) __bf16 As[2][128][40];   // [m][k] of x tile
    __shared__ __align__(16) __bf16 Bs[2][128][40];   // [col][k] of W_ih tile
    const int g0 = blockIdx.x * 128;
    const int R0 = blockIdx.y * 128;
    const int tid = threadIdx.x, w = tid >> 5, lane = tid & 31;

    v8f acc[8];
#pragma unroll
    for (int i = 0; i < 8; ++i) acc[i] = 0.0f;

    auto issue = [&](int b, int k0) {
#pragma unroll
        for (int i = 0; i < 4; ++i) {
            int ch = tid + 256 * i;
            if (ch < 512) {               // A tile: 128 rows x 32k, 4 chunks/row
                int row = ch >> 2, q = (ch & 3) * 8;
                async_cp16(LDSOFF(&As[b][row][q]),
                           xg + (size_t)(R0 + row) * PP + k0 + q);
            } else {                      // B tile: 128 cols x 32k
                int bc = ch - 512, row = bc >> 2, q = (bc & 3) * 8;
                async_cp16(LDSOFF(&Bs[b][row][q]),
                           wihb + (size_t)(g0 + row) * PP + k0 + q);
            }
        }
    };

    issue(0, 0);
    const int KT = PP / 32;
    for (int kt = 0; kt < KT; ++kt) {
        int b = kt & 1;
        if (kt + 1 < KT) {
            issue(b ^ 1, (kt + 1) * 32);
            asm volatile("s_wait_asynccnt 0x4" ::: "memory");  // tile kt done
        } else {
            asm volatile("s_wait_asynccnt 0x0" ::: "memory");
        }
        __syncthreads();

        v16bf a = frag_ld(&As[b][16 * w + (lane & 15)][0], lane);
#pragma unroll
        for (int ct = 0; ct < 8; ++ct) {
            v16bf bb = frag_ld(&Bs[b][ct * 16 + (lane & 15)][0], lane);
            acc[ct] = __builtin_amdgcn_wmma_f32_16x16x32_bf16(
                false, a, false, bb, (short)0, acc[ct], false, false);
        }
        __syncthreads();
    }

    // epilogue: D layout -> (row, gate col), add bias, store bf16
    const int mhi = (lane >> 4) * 8;
#pragma unroll
    for (int ct = 0; ct < 8; ++ct) {
        int gcol = g0 + ct * 16 + (lane & 15);
        float bias = bih[gcol];
#pragma unroll
        for (int rr = 0; rr < 8; ++rr) {
            size_t R = (size_t)(R0 + 16 * w + mhi + rr);
            gx[R * GG + gcol] = (__bf16)(acc[ct][rr] + bias);
        }
    }
}

// ================= Kernel 4: one GRU timestep (fused 3-gate GEMM + gate math)
// grid = (HH/64, NN/128); block = 256 (8 waves). M=128 rows(n), 64 cols(c), K=512.
// Double-buffered LDS fed by async copies (5 per wave per tile, uniform).
__global__ void k_gru(int t,
                      const float*  __restrict__ h_prev,
                      const __bf16* __restrict__ hb_prev,
                      float*  __restrict__ h_next,
                      __bf16* __restrict__ hb_next,
                      const __bf16* __restrict__ gx,
                      const __bf16* __restrict__ whhb,
                      const float* __restrict__ bhh,
                      float* __restrict__ out) {
    __shared__ __align__(16) __bf16 As[2][128][40];       // h tile [n][k]
    __shared__ __align__(16) __bf16 Bs[2][3][64][40];     // W_hh tiles [g][col][k]
    const int c0 = blockIdx.x * 64;
    const int n0 = blockIdx.y * 128;
    const int tid = threadIdx.x, w = tid >> 5, lane = tid & 31;

    v8f acc[3][4];
#pragma unroll
    for (int g = 0; g < 3; ++g)
#pragma unroll
        for (int c = 0; c < 4; ++c) acc[g][c] = 0.0f;

    auto issue = [&](int b, int k0) {
#pragma unroll
        for (int i = 0; i < 5; ++i) {
            int ch = tid + 256 * i;
            if (ch < 512) {               // A tile: 128 rows x 32k
                int row = ch >> 2, q = (ch & 3) * 8;
                async_cp16(LDSOFF(&As[b][row][q]),
                           hb_prev + (size_t)(n0 + row) * HH + k0 + q);
            } else {                      // B tiles: 3 gates x 64 cols x 32k
                int bc = ch - 512, row = bc >> 2, q = (bc & 3) * 8;
                int g = row >> 6, c = row & 63;
                async_cp16(LDSOFF(&Bs[b][g][c][q]),
                           whhb + (size_t)(g * HH + c0 + c) * HH + k0 + q);
            }
        }
    };

    issue(0, 0);
    const int KT = HH / 32;
    for (int kt = 0; kt < KT; ++kt) {
        int b = kt & 1;
        if (kt + 1 < KT) {
            issue(b ^ 1, (kt + 1) * 32);
            asm volatile("s_wait_asynccnt 0x5" ::: "memory");  // tile kt done
        } else {
            asm volatile("s_wait_asynccnt 0x0" ::: "memory");
        }
        __syncthreads();

        v16bf a = frag_ld(&As[b][16 * w + (lane & 15)][0], lane);
#pragma unroll
        for (int g = 0; g < 3; ++g)
#pragma unroll
            for (int ct = 0; ct < 4; ++ct) {
                v16bf bb = frag_ld(&Bs[b][g][ct * 16 + (lane & 15)][0], lane);
                acc[g][ct] = __builtin_amdgcn_wmma_f32_16x16x32_bf16(
                    false, a, false, bb, (short)0, acc[g][ct], false, false);
            }
        __syncthreads();
    }

    // epilogue: gate math, ping-pong h update, raw output
    const int mhi = (lane >> 4) * 8;
#pragma unroll
    for (int ct = 0; ct < 4; ++ct) {
        int c = c0 + ct * 16 + (lane & 15);
        float br = bhh[c], bu = bhh[HH + c], bn = bhh[2 * HH + c];
#pragma unroll
        for (int rr = 0; rr < 8; ++rr) {
            int n = n0 + 16 * w + mhi + rr;
            size_t Rg = ((size_t)t * NN + n) * GG;
            float ghr = acc[0][ct][rr] + br;
            float ghu = acc[1][ct][rr] + bu;
            float ghn = acc[2][ct][rr] + bn;
            float gxr = (float)gx[Rg + c];
            float gxu = (float)gx[Rg + HH + c];
            float gxn = (float)gx[Rg + 2 * HH + c];
            float rg = 1.0f / (1.0f + __expf(-(gxr + ghr)));
            float ug = 1.0f / (1.0f + __expf(-(gxu + ghu)));
            float ng = tanhf(gxn + rg * ghn);
            float hold = h_prev[(size_t)n * HH + c];
            float hnew = (1.0f - ug) * ng + ug * hold;
            h_next[(size_t)n * HH + c]  = hnew;
            hb_next[(size_t)n * HH + c] = (__bf16)hnew;
            out[((size_t)t * NN + n) * HH + c] = hnew;
        }
    }
}

// ================= Kernel 5: in-place norm-clamp + LayerNorm, one wave/row
__global__ void k_outln(float* __restrict__ out, const float* __restrict__ lg,
                        const float* __restrict__ lb) {
    int wave = threadIdx.x >> 5, lane = threadIdx.x & 31;
    size_t row = (size_t)blockIdx.x * 8 + wave;
    float* p = out + row * HH;
    float v[16], s = 0.0f, s2 = 0.0f;
#pragma unroll
    for (int j = 0; j < 16; ++j) {
        int c = lane + 32 * j;
        v[j] = p[c]; s += v[j]; s2 = fmaf(v[j], v[j], s2);
    }
    s = wred32(s); s2 = wred32(s2);
    float norm = sqrtf(s2);
    float sc   = fminf(5.0f / fmaxf(norm, 1e-12f), 1.0f);
    float mean = sc * s * (1.0f / HH);
    float var  = sc * sc * s2 * (1.0f / HH) - mean * mean;
    float rstd = rsqrtf(var + 1e-5f);
#pragma unroll
    for (int j = 0; j < 16; ++j) {
        int c = lane + 32 * j;
        p[c] = (sc * v[j] - mean) * rstd * lg[c] + lb[c];
    }
}

// ================= launch =================
extern "C" void kernel_launch(void* const* d_in, const int* in_sizes, int n_in,
                              void* d_out, int out_size, void* d_ws, size_t ws_size,
                              hipStream_t stream) {
    const float* z      = (const float*)d_in[0];
    const float* tpe    = (const float*)d_in[1];
    const float* W_proj = (const float*)d_in[2];
    const float* b_proj = (const float*)d_in[3];
    const float* ln_in_g = (const float*)d_in[4];
    const float* ln_in_b = (const float*)d_in[5];
    const float* W_ih   = (const float*)d_in[6];
    const float* b_ih   = (const float*)d_in[7];
    const float* W_hh   = (const float*)d_in[8];
    const float* b_hh   = (const float*)d_in[9];
    const float* ln_out_g = (const float*)d_in[10];
    const float* ln_out_b = (const float*)d_in[11];
    float* out = (float*)d_out;

    char* ws = (char*)d_ws;
    size_t off = 0;
    auto alloc = [&](size_t bytes) -> void* {
        void* p = ws + off; off += (bytes + 255) & ~(size_t)255; return p;
    };
    float*  az   = (float*) alloc((size_t)NN * PP * 4);
    float*  at   = (float*) alloc((size_t)TT * PP * 4);
    __bf16* xg   = (__bf16*)alloc((size_t)TT * NN * PP * 2);    // 512 MB
    __bf16* gx   = (__bf16*)alloc((size_t)TT * NN * GG * 2);    // 1536 MB
    float*  h0   = (float*) alloc((size_t)NN * HH * 4);
    float*  h1   = (float*) alloc((size_t)NN * HH * 4);
    __bf16* hb0  = (__bf16*)alloc((size_t)NN * HH * 2);
    __bf16* hb1  = (__bf16*)alloc((size_t)NN * HH * 2);
    __bf16* wihb = (__bf16*)alloc((size_t)GG * PP * 2);
    __bf16* whhb = (__bf16*)alloc((size_t)GG * HH * 2);

    hipMemsetAsync(h0, 0, (size_t)NN * HH * 4, stream);
    hipMemsetAsync(hb0, 0, (size_t)NN * HH * 2, stream);

    // one-time bf16 weight conversion
    int cvtThreads = (2 * GG * PP) / 4;
    k_cvtw<<<cvtThreads / 256, 256, 0, stream>>>(W_ih, W_hh, wihb, whhb);

    int totProj = NN * PP + TT * PP;
    k_proj<<<(totProj + 255) / 256, 256, 0, stream>>>(z, tpe, W_proj, b_proj, az, at);

    k_xln<<<(TT * NN) / 8, 256, 0, stream>>>(az, at, ln_in_g, ln_in_b, xg);

    dim3 ggx(GG / 128, (TT * NN) / 128);
    k_gx<<<ggx, 256, 0, stream>>>(xg, wihb, b_ih, gx);

    float *hp = h0, *hn = h1;
    __bf16 *hbp = hb0, *hbn = hb1;
    dim3 ggru(HH / 64, NN / 128);
    for (int t = 0; t < TT; ++t) {
        k_gru<<<ggru, 256, 0, stream>>>(t, hp, hbp, hn, hbn, gx, whhb, b_hh, out);
        float* tf = hp; hp = hn; hn = tf;
        __bf16* tb = hbp; hbp = hbn; hbn = tb;
    }

    k_outln<<<(TT * NN) / 8, 256, 0, stream>>>(out, ln_out_g, ln_out_b);
}
